// CRF_34583076668117
// MI455X (gfx1250) — matrix operand order repaired and checked by reading, not compiled, not run on hardware
//
#include <hip/hip_runtime.h>

// CRF log-likelihood, restructured so the forward-algorithm inner step is a
// linear-space GEMM executed on the CDNA5 matrix pipe (v_wmma_f32_16x16x32_f16).

typedef __attribute__((ext_vector_type(16))) _Float16 v16h;
typedef __attribute__((ext_vector_type(8)))  float    v8f;

#define B_SZ   2048
#define T_SZ   512
#define K_SZ   33
#define NT     3          // 3 N-tiles of 16 -> 48 padded "to" states
#define NSLOT  17         // states handled per lane (lo lanes get state 32)
#define ROWS   16         // batch rows per wave (WMMA M)
#define SSTR   52         // LDS row stride (floats), 48 used + pad

// ---------------------------------------------------------------------------
// Denominator: log partition via forward algorithm, WMMA inner step.
// One wave per 16 batch rows; serial loop over T.
// ---------------------------------------------------------------------------
__global__ __launch_bounds__(32) void crf_forward_wmma(
    const float* __restrict__ em,        // [B,T,K]
    const unsigned char* __restrict__ mask,  // [B,T]
    const float* __restrict__ startT,    // [K]
    const float* __restrict__ endT,      // [K]
    const float* __restrict__ trans,     // [K,K]
    float* __restrict__ logz_out)        // [B]
{
    __shared__ float smem[ROWS * SSTR];

    const int  lane = threadIdx.x;        // 0..31
    const bool hi   = lane >= 16;
    const int  row  = lane & 15;          // batch row within the 16-row tile
    const int  b    = blockIdx.x * ROWS + row;

    // slot -> state index, matching the 16-bit WMMA A-matrix half layout:
    // lanes 0-15 : halves 0..7 = K 0..7,  halves 8..15 = K 16..23, slot16 = K 32
    // lanes16-31 : halves 0..7 = K 8..15, halves 8..15 = K 24..31, slot16 = pad
    int jmap[NSLOT];
#pragma unroll
    for (int s = 0; s < 16; ++s) jmap[s] = (s < 8 ? s : s + 8) + (hi ? 8 : 0);
    jmap[16] = hi ? K_SZ : 32;            // K_SZ == invalid sentinel

    // ---- resident B operand: E[k][j] = exp(trans[k][j]), zero padded ------
    // B 32x16 f16 layout: lane L holds column N = L&15, halves = K in jmap (table) order.
    v16h Bt[2][NT];
#pragma unroll
    for (int kc = 0; kc < 2; ++kc) {
#pragma unroll
        for (int nt = 0; nt < NT; ++nt) {
#pragma unroll
            for (int h = 0; h < 16; ++h) {
                const int k = kc * 32 + (h < 8 ? h : h + 8) + (hi ? 8 : 0);
                const int j = nt * 16 + (lane & 15);
                float v = 0.0f;
                if (k < K_SZ && j < K_SZ) v = __expf(trans[k * K_SZ + j]);
                Bt[kc][nt][h] = (_Float16)v;
            }
        }
    }

    float expEnd[NSLOT];
#pragma unroll
    for (int s = 0; s < NSLOT; ++s)
        expEnd[s] = (jmap[s] < K_SZ) ? __expf(endT[jmap[s]]) : 0.0f;

    const float* emrow = em + (size_t)b * T_SZ * K_SZ;

    // ---- t = 0: alpha0 = start + e0; carry (M, p=exp(alpha-M)) ------------
    float p[NSLOT], a0[NSLOT];
    float c = -3.0e38f;
#pragma unroll
    for (int s = 0; s < NSLOT; ++s) {
        if (jmap[s] < K_SZ) {
            a0[s] = startT[jmap[s]] + emrow[jmap[s]];
            c = fmaxf(c, a0[s]);
        } else a0[s] = -3.0e38f;
    }
    c = fmaxf(c, __shfl_xor(c, 16, 32));   // combine lane pair -> max over 33
    float M = c;
#pragma unroll
    for (int s = 0; s < NSLOT; ++s)
        p[s] = (jmap[s] < K_SZ) ? __expf(a0[s] - c) : 0.0f;

    // pack p into the f16 A operand (two K-chunks: states 0..31 / 32..63)
    v16h A0, A1;
#pragma unroll
    for (int h = 0; h < 16; ++h) { A0[h] = (_Float16)p[h]; A1[h] = (_Float16)0.0f; }
    A1[0] = (_Float16)p[16];               // state 32 (hi lanes: padding == 0)

    // ---- serial recurrence over T ----------------------------------------
    for (int t = 1; t < T_SZ; ++t) {
        const float* e_t = emrow + (size_t)t * K_SZ;
        if (t + 1 < T_SZ) __builtin_prefetch(e_t + K_SZ, 0, 1);  // next step

        float e[NSLOT];
        float cl = -3.0e38f;
#pragma unroll
        for (int s = 0; s < NSLOT; ++s) {
            if (jmap[s] < K_SZ) { e[s] = e_t[jmap[s]]; cl = fmaxf(cl, e[s]); }
            else e[s] = 0.0f;
        }
        const float cc = fmaxf(cl, __shfl_xor(cl, 16, 32));

        // s[b,j] = sum_i p[b,i] * E[i,j] : 16x48 result, 6 WMMAs
#pragma unroll
        for (int nt = 0; nt < NT; ++nt) {
            v8f acc = {};
            acc = __builtin_amdgcn_wmma_f32_16x16x32_f16(
                      false, A0, false, Bt[0][nt], (short)0, acc, false, false);
            acc = __builtin_amdgcn_wmma_f32_16x16x32_f16(
                      false, A1, false, Bt[1][nt], (short)0, acc, false, false);
            // D layout: VGPR r on lane L -> (M = r + (L>=16?8:0), N = nt*16 + (L&15))
#pragma unroll
            for (int r = 0; r < 8; ++r)
                smem[(r + (hi ? 8 : 0)) * SSTR + nt * 16 + (lane & 15)] = acc[r];
        }
        __syncthreads();   // single-wave WG: orders DS store -> load

        const unsigned char mk = mask[(size_t)b * T_SZ + t];

        // q[j] = s[j] * exp(e[j] - cc);   renormalize so max(p') == 1
        float q[NSLOT];
        float rl = 0.0f;
#pragma unroll
        for (int s = 0; s < NSLOT; ++s) {
            float sv = (jmap[s] < K_SZ) ? smem[row * SSTR + jmap[s]] : 0.0f;
            q[s] = sv * __expf(e[s] - cc);
            rl = fmaxf(rl, q[s]);
        }
        const float r    = fmaxf(rl, __shfl_xor(rl, 16, 32));
        const float rinv = __builtin_amdgcn_rcpf(r);
        const float Mn   = M + cc + __logf(r);
        if (mk) {
            M = Mn;
#pragma unroll
            for (int s = 0; s < NSLOT; ++s) p[s] = q[s] * rinv;
        }
        // repack A operand for the next step
#pragma unroll
        for (int h = 0; h < 16; ++h) A0[h] = (_Float16)p[h];
        A1[0] = (_Float16)p[16];
        __syncthreads();   // protect smem before next iteration's stores
    }

    // logZ = M + log( sum_j p[j] * exp(end[j]) )
    float zl = 0.0f;
#pragma unroll
    for (int s = 0; s < NSLOT; ++s) zl += p[s] * expEnd[s];
    const float z = zl + __shfl_xor(zl, 16, 32);
    if (!hi) logz_out[b] = M + __logf(z);
}

// ---------------------------------------------------------------------------
// Numerator: score of the given tag path. One thread per sequence.
// ---------------------------------------------------------------------------
__global__ __launch_bounds__(256) void crf_numerator(
    const float* __restrict__ em, const int* __restrict__ tags,
    const unsigned char* __restrict__ mask,
    const float* __restrict__ startT, const float* __restrict__ endT,
    const float* __restrict__ trans, float* __restrict__ score_out)
{
    const int b = blockIdx.x * blockDim.x + threadIdx.x;
    if (b >= B_SZ) return;
    const float* emrow = em + (size_t)b * T_SZ * K_SZ;
    int   prev  = tags[(size_t)b * T_SZ];
    float score = startT[prev] + emrow[prev];
    for (int t = 1; t < T_SZ; ++t) {
        const int tg = tags[(size_t)b * T_SZ + t];
        const unsigned char mk = mask[(size_t)b * T_SZ + t];
        const float step = trans[prev * K_SZ + tg] + emrow[(size_t)t * K_SZ + tg];
        if (mk) { score += step; prev = tg; }
    }
    score_out[b] = score + endT[prev];
}

// ---------------------------------------------------------------------------
// Final reduction: mean over B of (score - logZ).
// ---------------------------------------------------------------------------
__global__ __launch_bounds__(256) void crf_reduce(
    const float* __restrict__ score, const float* __restrict__ logz,
    float* __restrict__ out)
{
    __shared__ float red[256];
    float acc = 0.0f;
    for (int b = threadIdx.x; b < B_SZ; b += 256) acc += score[b] - logz[b];
    red[threadIdx.x] = acc;
    __syncthreads();
    for (int s = 128; s > 0; s >>= 1) {
        if (threadIdx.x < s) red[threadIdx.x] += red[threadIdx.x + s];
        __syncthreads();
    }
    if (threadIdx.x == 0) out[0] = red[0] * (1.0f / (float)B_SZ);
}

// ---------------------------------------------------------------------------
extern "C" void kernel_launch(void* const* d_in, const int* in_sizes, int n_in,
                              void* d_out, int out_size, void* d_ws, size_t ws_size,
                              hipStream_t stream)
{
    const float*         em     = (const float*)d_in[0];
    const int*           tags   = (const int*)d_in[1];
    const unsigned char* mask   = (const unsigned char*)d_in[2];
    const float*         startT = (const float*)d_in[3];
    const float*         endT   = (const float*)d_in[4];
    const float*         trans  = (const float*)d_in[5];

    float* ws    = (float*)d_ws;
    float* score = ws;            // [B]
    float* logz  = ws + B_SZ;     // [B]

    crf_numerator<<<B_SZ / 256, 256, 0, stream>>>(em, tags, mask, startT, endT,
                                                  trans, score);
    crf_forward_wmma<<<B_SZ / ROWS, 32, 0, stream>>>(em, mask, startT, endT,
                                                     trans, logz);
    crf_reduce<<<1, 256, 0, stream>>>(score, logz, (float*)d_out);
}
